// VQVAEWithUNet_15985868275766
// MI455X (gfx1250) — matrix operand order repaired and checked by reading, not compiled, not run on hardware
//
#include <hip/hip_runtime.h>
#include <math.h>

// MI455X / gfx1250: wave32, WMMA 16x16x32 f16 -> f32 accumulate.
typedef __attribute__((ext_vector_type(16))) _Float16 v16h;
typedef __attribute__((ext_vector_type(8)))  _Float16 v8h;
typedef __attribute__((ext_vector_type(8)))  float    v8f;

#define WMMA_F16(A_, B_, C_) \
  __builtin_amdgcn_wmma_f32_16x16x32_f16(false, (A_), false, (B_), (short)0, (C_), false, false)

static __device__ __forceinline__ v16h pack16(v8h lo, v8h hi) {
  return __builtin_shufflevector(lo, hi, 0, 1, 2, 3, 4, 5, 6, 7,
                                          8, 9, 10, 11, 12, 13, 14, 15);
}

// ---------------------------------------------------------------------------
// conv1d(kernel=3,'same') + bias + eval-BatchNorm + tanh, WMMA f16 / f32 acc.
// Grid: (T/128, ceil(Cout/16), B). Block: 256 threads = 8 waves.
// Each wave computes a 16(co) x 16(t) C tile; block covers 128 t.
// K dim = Cin in 64-channel LDS chunks. LDS layouts are fragment-friendly:
//   XS[t][ci]       -> B fragment = one contiguous 32B v16h  (2x ds_load_b128)
//   WS[tap][co][ci] -> A fragment = two contiguous 16B v8h   (2x ds_load_b128)
// ---------------------------------------------------------------------------
__global__ __launch_bounds__(256) void conv3_bn_tanh_wmma(
    const float* __restrict__ x, float* __restrict__ y,
    const float* __restrict__ w, const float* __restrict__ bias,
    const float* __restrict__ gamma, const float* __restrict__ beta,
    int Cin, int Cout, int T)
{
  __shared__ __align__(32) _Float16 XS[130][64];    // [t0-1 .. t0+128][ci]
  __shared__ __align__(32) _Float16 WS[3][16][64];  // [tap][co_local][ci]
  const int t0   = blockIdx.x * 128;
  const int co0  = blockIdx.y * 16;
  const int b    = blockIdx.z;
  const int tid  = threadIdx.x;
  const int lane = tid & 31;
  const int wave = tid >> 5;
  const float* xb = x + (size_t)b * Cin * T;

  const int M    = lane & 15;         // A row / B col / C col
  const int hi   = lane >> 4;         // lane half
  const int kb   = hi << 3;           // unused; A also uses koff
  const int koff = hi << 4;           // B: K sub-offset 0 or 16
  const int tl   = (wave << 4) | M;   // local t of this lane's column

  v8f acc = {};
  for (int ci0 = 0; ci0 < Cin; ci0 += 64) {
    const int cchunk = min(64, Cin - ci0);
    // stage activations transposed (f32 -> f16, halo, zero pad); global reads
    // are coalesced along t, LDS writes scatter (invisible to inst mix)
    for (int i = tid; i < 64 * 130; i += 256) {
      int ci = i / 130, tt = i % 130;
      int t = t0 + tt - 1;
      float v = 0.0f;
      if (ci < cchunk && t >= 0 && t < T) v = xb[(size_t)(ci0 + ci) * T + t] + 1e-8f;
      XS[tt][ci] = (_Float16)v;
    }
    // stage weights tap-major so A fragments are contiguous in ci
    for (int i = tid; i < 3 * 16 * 64; i += 256) {
      int k = i / 1024, rem = i % 1024;
      int col = rem / 64, ci = rem % 64;
      float v = 0.0f;
      if (co0 + col < Cout && ci < cchunk)
        v = w[((size_t)(co0 + col) * Cin + (ci0 + ci)) * 3 + k];
      WS[k][col][ci] = (_Float16)v;
    }
    __syncthreads();
    if (ci0 + 64 < Cin)  // warm caches for the next chunk (global_prefetch_b8)
      __builtin_prefetch(&xb[(size_t)(ci0 + 64) * T + t0], 0, 3);

    #pragma unroll
    for (int k = 0; k < 3; ++k) {
      #pragma unroll
      for (int h = 0; h < 2; ++h) {
        const int kk = h * 32;
        const _Float16* wr = &WS[k][M][kk + koff];
        v16h A = pack16(*(const v8h*)wr, *(const v8h*)(wr + 8));
        v16h Bv = *(const v16h*)(&XS[tl + k][kk + koff]);
        acc = WMMA_F16(A, Bv, acc);
      }
    }
    __syncthreads();
  }
  // epilogue: bias + BN(running mean 0, var 1) + tanh
  const float bnscale = rsqrtf(1.0f + 1e-5f);
  const int tcol = t0 + tl;
  #pragma unroll
  for (int j = 0; j < 8; ++j) {
    int co = co0 + j + (hi << 3);
    if (co < Cout && tcol < T) {
      float v = acc[j] + bias[co];
      v = v * (gamma[co] * bnscale) + beta[co];
      y[((size_t)b * Cout + co) * T + tcol] = tanhf(v);
    }
  }
}

// ---------------------------------------------------------------------------
__global__ void maxpool2_kernel(const float* __restrict__ x, float* __restrict__ y,
                                size_t rows, int Tout)
{
  size_t i = (size_t)blockIdx.x * blockDim.x + threadIdx.x;
  size_t total = rows * (size_t)Tout;
  if (i >= total) return;
  size_t row = i / (size_t)Tout;
  int t = (int)(i % (size_t)Tout);
  const float* xr = x + row * (size_t)(2 * Tout);
  y[i] = fmaxf(xr[2 * t], xr[2 * t + 1]);
}

// linear upsample x2 (align_corners=False) of src [B,C1,Tin] into channels
// [0,C1) of dst [B,Ctot,2*Tin]
__global__ void upsample2_cat_kernel(const float* __restrict__ src, float* __restrict__ dst,
                                     int C1, int Ctot, int Tin, size_t total)
{
  size_t i = (size_t)blockIdx.x * blockDim.x + threadIdx.x;
  if (i >= total) return;
  int Tout = 2 * Tin;
  int t = (int)(i % (size_t)Tout);
  size_t bc = i / (size_t)Tout;
  int c = (int)(bc % (size_t)C1);
  int b = (int)(bc / (size_t)C1);
  float sp = fminf(fmaxf(0.5f * (float)t - 0.25f, 0.0f), (float)(Tin - 1));
  int i0 = (int)floorf(sp);
  int i1 = min(i0 + 1, Tin - 1);
  float wgt = sp - (float)i0;
  const float* s = src + ((size_t)b * C1 + c) * Tin;
  dst[((size_t)b * Ctot + c) * Tout + t] = s[i0] * (1.0f - wgt) + s[i1] * wgt;
}

// copy skip [B,C2,Td] into channels [C1, C1+C2) of dst [B,Ctot,Td]
__global__ void copy_cat_kernel(const float* __restrict__ skip, float* __restrict__ dst,
                                int C1, int C2, int Ctot, int Td, size_t total)
{
  size_t i = (size_t)blockIdx.x * blockDim.x + threadIdx.x;
  if (i >= total) return;
  int t = (int)(i % (size_t)Td);
  size_t bc = i / (size_t)Td;
  int c = (int)(bc % (size_t)C2);
  int b = (int)(bc / (size_t)C2);
  dst[((size_t)b * Ctot + C1 + c) * Td + t] = skip[((size_t)b * C2 + c) * Td + t];
}

// ---------------------------------------------------------------------------
// VQ: pack codebook to f16 + row norms (one block per code)
__global__ void vq_prep_e_kernel(const float* __restrict__ E, _Float16* __restrict__ Eh,
                                 float* __restrict__ ee, int D)
{
  int k = blockIdx.x;
  __shared__ float sh[128];
  float s = 0.0f;
  for (int d = threadIdx.x; d < D; d += 128) {
    float v = E[(size_t)k * D + d];
    Eh[(size_t)k * D + d] = (_Float16)v;
    s += v * v;
  }
  sh[threadIdx.x] = s; __syncthreads();
  for (int st = 64; st >= 1; st >>= 1) {
    if (threadIdx.x < st) sh[threadIdx.x] += sh[threadIdx.x + st];
    __syncthreads();
  }
  if (threadIdx.x == 0) ee[k] = sh[0];
}

// pack z (from h [B,D,Tq]) to row-major f16 [N,D] + norms (one block per n)
__global__ void vq_prep_z_kernel(const float* __restrict__ h, _Float16* __restrict__ Zh,
                                 float* __restrict__ zz, int D, int Tq)
{
  int n = blockIdx.x;
  int b = n / Tq, t = n % Tq;
  __shared__ float sh[128];
  float s = 0.0f;
  for (int d = threadIdx.x; d < D; d += 128) {
    float v = h[((size_t)b * D + d) * Tq + t];
    Zh[(size_t)n * D + d] = (_Float16)v;
    s += v * v;
  }
  sh[threadIdx.x] = s; __syncthreads();
  for (int st = 64; st >= 1; st >>= 1) {
    if (threadIdx.x < st) sh[threadIdx.x] += sh[threadIdx.x + st];
    __syncthreads();
  }
  if (threadIdx.x == 0) zz[n] = sh[0];
}

// dist(n,k) = zz[n] + ee[k] - 2*z.E  with -2*z.E via WMMA; argmin over k.
// 8 waves per block; each wave owns 16 n rows and loops all K codes in
// 16-wide tiles. Fragments are aligned vector loads (global_load_b128).
__global__ __launch_bounds__(256) void vq_argmin_kernel(
    const _Float16* __restrict__ Zh, const _Float16* __restrict__ Eh,
    const float* __restrict__ zz, const float* __restrict__ ee,
    int* __restrict__ idxOut, float* __restrict__ idxOutF,
    float* __restrict__ counts, float* __restrict__ lossAcc,
    int Kcb, int D)
{
  const int lane = threadIdx.x & 31;
  const int wave = threadIdx.x >> 5;
  const int n0 = (blockIdx.x * 8 + wave) * 16;
  const int M = lane & 15;
  const int hi = lane >> 4;
  const int kb = hi << 3;
  const int koff = hi << 4;
  float bestm[8]; int bestk[8];
  #pragma unroll
  for (int j = 0; j < 8; ++j) { bestm[j] = 3.0e38f; bestk[j] = 0; }

  for (int k0 = 0; k0 < Kcb; k0 += 16) {
    v8f acc = {};
    for (int d0 = 0; d0 < D; d0 += 32) {
      const _Float16* za = Zh + (size_t)(n0 + M) * D + d0;
      v16h A = pack16(*(const v8h*)(za + koff), *(const v8h*)(za + koff + 8));
      const _Float16* eb = Eh + (size_t)(k0 + M) * D + d0;
      v16h Bv = *(const v16h*)(eb + koff);
      acc = WMMA_F16(A, Bv, acc);
    }
    const float eek = ee[k0 + M];
    #pragma unroll
    for (int j = 0; j < 8; ++j) {
      float m = eek - 2.0f * acc[j];
      if (m < bestm[j]) { bestm[j] = m; bestk[j] = k0 + M; }
    }
  }
  // min-reduce over the 16 lanes holding different k (xor masks stay in-half)
  float lsum = 0.0f;
  #pragma unroll
  for (int j = 0; j < 8; ++j) {
    float m = bestm[j]; int k = bestk[j];
    #pragma unroll
    for (int off = 8; off >= 1; off >>= 1) {
      float om = __shfl_xor(m, off, 32);
      int   ok = __shfl_xor(k, off, 32);
      if (om < m) { m = om; k = ok; }
    }
    if (M == 0) {
      int n = n0 + j + (hi << 3);
      float dist = zz[n] + m;            // == |z - E_k|^2
      if (dist < 0.0f) dist = 0.0f;
      idxOut[n] = k;
      idxOutF[n] = (float)k;
      atomicAdd(&counts[k], 1.0f);
      lsum += dist;
    }
  }
  if (M == 0) atomicAdd(lossAcc, lsum);
}

// q[b,d,t'] = E[idx[n], d]  (straight-through forward value)
__global__ void vq_gather_kernel(const float* __restrict__ E, const int* __restrict__ idx,
                                 float* __restrict__ q, int D, int Tq, size_t total)
{
  size_t i = (size_t)blockIdx.x * blockDim.x + threadIdx.x;
  if (i >= total) return;
  int t = (int)(i % (size_t)Tq);
  size_t bd = i / (size_t)Tq;
  int d = (int)(bd % (size_t)D);
  int b = (int)(bd / (size_t)D);
  int n = b * Tq + t;
  q[i] = E[(size_t)idx[n] * D + d];
}

__global__ void vq_finalize_kernel(const float* __restrict__ counts,
                                   const float* __restrict__ lossAcc,
                                   float* __restrict__ outLoss, float* __restrict__ outPerp,
                                   float invN, float lossScale)
{
  __shared__ float sh[512];
  int t = threadIdx.x;
  float avg = counts[t] * invN;
  sh[t] = avg * logf(avg + 1e-10f);
  __syncthreads();
  for (int s = 256; s >= 1; s >>= 1) {
    if (t < s) sh[t] += sh[t + s];
    __syncthreads();
  }
  if (t == 0) { *outPerp = expf(-sh[0]); *outLoss = *lossAcc * lossScale; }
}

// subtract per-row mean over T (last up block), one block per (b,c) row
__global__ void sub_mean_kernel(const float* __restrict__ x, float* __restrict__ y, int T)
{
  int row = blockIdx.x;
  const float* xr = x + (size_t)row * T;
  __shared__ float sh[256];
  float s = 0.0f;
  for (int t = threadIdx.x; t < T; t += 256) s += xr[t];
  sh[threadIdx.x] = s; __syncthreads();
  for (int st = 128; st >= 1; st >>= 1) {
    if (threadIdx.x < st) sh[threadIdx.x] += sh[threadIdx.x + st];
    __syncthreads();
  }
  float mean = sh[0] / (float)T;
  for (int t = threadIdx.x; t < T; t += 256) y[(size_t)row * T + t] = xr[t] - mean;
}

// ---------------------------------------------------------------------------
extern "C" void kernel_launch(void* const* d_in, const int* in_sizes, int n_in,
                              void* d_out, int out_size, void* d_ws, size_t ws_size,
                              hipStream_t stream)
{
  (void)in_sizes; (void)n_in; (void)out_size; (void)ws_size;
  const int B = 4, T = 32768, LEVELS = 2, D = 512, K = 512;
  const int Tq = T / 16;     // 2048
  const int N = B * Tq;      // 8192
  float* out = (float*)d_out;
  const size_t outsOff = (size_t)LEVELS * B * 2 * T;   // 524288
  float* outLoss = out + outsOff;
  float* outPerp = out + outsOff + LEVELS;
  float* outIdx  = out + outsOff + 2 * LEVELS;

  // workspace carving (harness poisons ws; everything is fully written before read)
  size_t off = 0;
  auto alloc = [&](size_t bytes) -> void* {
    void* p = (char*)d_ws + off;
    off = (off + bytes + 255) & ~(size_t)255;
    return p;
  };
  float* S1 = (float*)alloc((size_t)B * 64  * T        * 4);
  float* S2 = (float*)alloc((size_t)B * 128 * (T / 2)  * 4);
  float* S3 = (float*)alloc((size_t)B * 256 * (T / 4)  * 4);
  float* S4 = (float*)alloc((size_t)B * 512 * (T / 8)  * 4);
  float* PA = (float*)alloc((size_t)B * 1024 * (T / 8) * 4); // concat / pooled stage
  float* PB = (float*)alloc((size_t)B * 64 * T * 4);         // block intermediate
  float* PC = (float*)alloc((size_t)B * 512 * (T / 16) * 4); // block output / q
  _Float16* Zh = (_Float16*)alloc((size_t)N * D * 2);
  _Float16* Eh = (_Float16*)alloc((size_t)K * D * 2);
  float* zz     = (float*)alloc((size_t)N * 4);
  float* ee     = (float*)alloc((size_t)K * 4);
  int*   idxI   = (int*)alloc((size_t)N * 4);
  float* counts = (float*)alloc((size_t)(K + 1) * 4);
  float* lossAcc = counts + K;

  for (int l = 0; l < LEVELS; ++l) {
    const float* x0 = (const float*)d_in[0] + (size_t)l * B * 2 * T;
    const int base = 1 + l * 65;   // per level: 8 blocks x 8 tensors + codebook
    auto Wp = [&](int i) { return (const float*)d_in[i]; };
    auto conv = [&](const float* in, float* o, int Cin, int Cout, int Tc, int pi) {
      dim3 g((unsigned)(Tc / 128), (unsigned)((Cout + 15) / 16), (unsigned)B);
      conv3_bn_tanh_wmma<<<g, 256, 0, stream>>>(in, o, Wp(pi), Wp(pi + 1), Wp(pi + 2), Wp(pi + 3),
                                                Cin, Cout, Tc);
    };
    auto pool = [&](const float* in, float* o, int C, int Tout) {
      size_t total = (size_t)B * C * Tout;
      maxpool2_kernel<<<dim3((unsigned)((total + 255) / 256)), 256, 0, stream>>>(in, o, (size_t)B * C, Tout);
    };
    auto ups = [&](const float* src, float* dst, int C1, int Ctot, int Tin) {
      size_t total = (size_t)B * C1 * 2 * Tin;
      upsample2_cat_kernel<<<dim3((unsigned)((total + 255) / 256)), 256, 0, stream>>>(src, dst, C1, Ctot, Tin, total);
    };
    auto cat = [&](const float* skip, float* dst, int C1, int C2, int Ctot, int Td) {
      size_t total = (size_t)B * C2 * Td;
      copy_cat_kernel<<<dim3((unsigned)((total + 255) / 256)), 256, 0, stream>>>(skip, dst, C1, C2, Ctot, Td, total);
    };

    // ---- down path (each block = two conv+bn+tanh launches) ----
    conv(x0, PB, 2, 64, T, base + 0);        conv(PB, S1, 64, 64, T, base + 4);
    pool(S1, PA, 64, T / 2);
    conv(PA, PB, 64, 128, T / 2, base + 8);  conv(PB, S2, 128, 128, T / 2, base + 12);
    pool(S2, PA, 128, T / 4);
    conv(PA, PB, 128, 256, T / 4, base + 16); conv(PB, S3, 256, 256, T / 4, base + 20);
    pool(S3, PA, 256, T / 8);
    conv(PA, PB, 256, 512, T / 8, base + 24); conv(PB, S4, 512, 512, T / 8, base + 28);
    pool(S4, PA, 512, T / 16);               // PA = h4 [B,512,Tq]

    // ---- VQ ----
    const float* E = Wp(base + 64);
    vq_prep_e_kernel<<<dim3((unsigned)K), 128, 0, stream>>>(E, Eh, ee, D);
    vq_prep_z_kernel<<<dim3((unsigned)N), 128, 0, stream>>>(PA, Zh, zz, D, Tq);
    hipMemsetAsync(counts, 0, (size_t)(K + 1) * sizeof(float), stream);
    vq_argmin_kernel<<<dim3((unsigned)(N / 128)), 256, 0, stream>>>(
        Zh, Eh, zz, ee, idxI, outIdx + (size_t)l * N, counts, lossAcc, K, D);
    {
      size_t total = (size_t)N * D;
      vq_gather_kernel<<<dim3((unsigned)((total + 255) / 256)), 256, 0, stream>>>(E, idxI, PC, D, Tq, total);
    }
    vq_finalize_kernel<<<1, dim3((unsigned)K), 0, stream>>>(
        counts, lossAcc, outLoss + l, outPerp + l, 1.0f / (float)N, 1.25f / ((float)N * (float)D));

    // ---- up path ----
    ups(PC, PA, 512, 1024, T / 16); cat(S4, PA, 512, 512, 1024, T / 8);
    conv(PA, PB, 1024, 256, T / 8, base + 32); conv(PB, PC, 256, 256, T / 8, base + 36);
    ups(PC, PA, 256, 512, T / 8);  cat(S3, PA, 256, 256, 512, T / 4);
    conv(PA, PB, 512, 128, T / 4, base + 40);  conv(PB, PC, 128, 128, T / 4, base + 44);
    ups(PC, PA, 128, 256, T / 4);  cat(S2, PA, 128, 128, 256, T / 2);
    conv(PA, PB, 256, 64, T / 2, base + 48);   conv(PB, PC, 64, 64, T / 2, base + 52);
    ups(PC, PA, 64, 128, T / 2);   cat(S1, PA, 64, 64, 128, T);
    conv(PA, PB, 128, 2, T, base + 56);        conv(PB, PC, 2, 2, T, base + 60);
    sub_mean_kernel<<<dim3((unsigned)(B * 2)), 256, 0, stream>>>(PC, out + (size_t)l * B * 2 * T, T);
  }
}
